// SeparableConv1d_17471926960196
// MI455X (gfx1250) — compile-verified, matrix operand order checked by reading, not run on hardware
//
#include <hip/hip_runtime.h>
#include <hip/hip_bf16.h>

// ---- problem constants (match reference) ----
#define BATCH   16
#define C_IN    512
#define C_OUT   512
#define K_DW    11
#define T_IN    2048
#define T_OUT   2038   // T_IN - K_DW + 1
#define T_PAD   2048   // padded N dimension for the GEMM (multiple of 128)

typedef __attribute__((ext_vector_type(16))) __bf16 v16bf;
typedef __attribute__((ext_vector_type(8)))  float  v8f;

union FragU {
    uint4  u4[2];   // two 16-byte global loads
    v16bf  v;       // WMMA operand view
};

__device__ __forceinline__ unsigned short f32_to_bf16(float f) {
    unsigned int u = __float_as_uint(f);
    // round-to-nearest-even
    u += 0x7FFFu + ((u >> 16) & 1u);
    return (unsigned short)(u >> 16);
}

// ---------------- kernel 1: pw_w f32 -> bf16 ----------------
__global__ __launch_bounds__(256) void cvt_w_kernel(const float* __restrict__ w,
                                                    unsigned short* __restrict__ wb,
                                                    int n) {
    int i = blockIdx.x * 256 + threadIdx.x;
    if (i < n) wb[i] = f32_to_bf16(w[i]);
}

// ---------------- kernel 2: depthwise conv -> Yt[b][t][c] (bf16) ----------------
// Block: 256 threads. Tile: 32 channels x 128 time-steps, transposed through LDS
// so x reads are coalesced along t and Yt writes are coalesced along c.
__global__ __launch_bounds__(256) void dw_kernel(const float* __restrict__ x,     // [B][C][T_IN]
                                                 const float* __restrict__ dww,   // [C][1][K]
                                                 const float* __restrict__ dwb,   // [C]
                                                 unsigned short* __restrict__ yt) // [B][T_PAD][C]
{
    __shared__ unsigned short tile[32][128 + 4];

    const int b  = blockIdx.z;
    const int c0 = blockIdx.y * 32;
    const int t0 = blockIdx.x * 128;
    const int tid = threadIdx.x;

    // phase 1: compute, lanes along t (coalesced x reads)
    {
        const int tl     = tid & 127;
        const int cstart = tid >> 7;       // 0..1
        const int t      = t0 + tl;
        #pragma unroll 1
        for (int cl = cstart; cl < 32; cl += 2) {
            const int c = c0 + cl;
            float acc = 0.0f;
            if (t < T_OUT) {
                const float* xp = x + ((size_t)b * C_IN + c) * T_IN + t;
                const float* wp = dww + c * K_DW;   // scalar (uniform) loads
                #pragma unroll
                for (int k = 0; k < K_DW; ++k) acc = fmaf(xp[k], wp[k], acc);
                acc += dwb[c];
            }
            tile[cl][tl] = f32_to_bf16(acc);
        }
    }
    __syncthreads();

    // phase 2: store, lanes along c (coalesced bf16 writes into transposed Yt)
    {
        const int cl     = tid & 31;
        const int tstart = tid >> 5;       // 0..7
        #pragma unroll 1
        for (int tl = tstart; tl < 128; tl += 8) {
            const int t = t0 + tl;
            if (t < T_OUT)
                yt[((size_t)b * T_PAD + t) * C_IN + (c0 + cl)] = tile[cl][tl];
        }
    }
}

// ---------------- kernel 3: pointwise GEMM via bf16 WMMA ----------------
// Out[b][o][t] = sum_c Wb[o][c] * Yt[b][t][c] + pw_b[o]
// Grid: (T_PAD/128, C_OUT/128, B); block 256 = 8 waves.
// Wave tile 64(M) x 32(N): 4x2 accumulators of 16x16 f32.
// Per-lane A/B fragments = two contiguous 16-byte global loads each
// (K = half*8 + [0..7] and 16 + half*8 + [0..7], contiguous bf16 in memory).
__global__ __launch_bounds__(256) void gemm_kernel(const unsigned short* __restrict__ wb,  // [C_OUT][C_IN] bf16
                                                   const unsigned short* __restrict__ yt,  // [B][T_PAD][C_IN] bf16
                                                   const float* __restrict__ pwb,          // [C_OUT]
                                                   float* __restrict__ out)                // [B][C_OUT][T_OUT]
{
    const int b   = blockIdx.z;
    const int o0  = blockIdx.y * 128;
    const int t0  = blockIdx.x * 128;
    const int tid  = threadIdx.x;
    const int wave = tid >> 5;
    const int lane = tid & 31;
    const int wm   = wave >> 2;            // 0..1 -> M offset 0/64
    const int wn   = wave & 3;             // 0..3 -> N offset 0/32/64/96
    const int mbase = o0 + wm * 64;
    const int nbase = t0 + wn * 32;
    const int lm   = lane & 15;            // row (A) / col (B,D) within 16
    const int half = lane >> 4;            // K-half selector

    const unsigned short* Bbase = yt + (size_t)b * T_PAD * C_IN;

    v8f acc[4][2];
    #pragma unroll
    for (int mi = 0; mi < 4; ++mi)
        #pragma unroll
        for (int ni = 0; ni < 2; ++ni)
            acc[mi][ni] = (v8f){0.f, 0.f, 0.f, 0.f, 0.f, 0.f, 0.f, 0.f};

    #pragma unroll 1
    for (int kk = 0; kk < C_IN; kk += 32) {
        FragU afr[4], bfr[2];
        #pragma unroll
        for (int mi = 0; mi < 4; ++mi) {
            const unsigned short* p =
                wb + (size_t)(mbase + mi * 16 + lm) * C_IN + kk + half * 8;
            afr[mi].u4[0] = *(const uint4*)(p);
            afr[mi].u4[1] = *(const uint4*)(p + 16);
        }
        #pragma unroll
        for (int ni = 0; ni < 2; ++ni) {
            const unsigned short* p =
                Bbase + (size_t)(nbase + ni * 16 + lm) * C_IN + kk + half * 8;
            bfr[ni].u4[0] = *(const uint4*)(p);
            bfr[ni].u4[1] = *(const uint4*)(p + 16);
        }
        // prefetch next K-step of the streamed operand (global_prefetch_b8)
        if (kk + 32 < C_IN)
            __builtin_prefetch(Bbase + (size_t)(nbase + lm) * C_IN + kk + 32, 0, 0);

        #pragma unroll
        for (int mi = 0; mi < 4; ++mi)
            #pragma unroll
            for (int ni = 0; ni < 2; ++ni)
                acc[mi][ni] = __builtin_amdgcn_wmma_f32_16x16x32_bf16(
                    /*neg_a=*/false, afr[mi].v,
                    /*neg_b=*/false, bfr[ni].v,
                    /*c_mod=*/(short)0, acc[mi][ni],
                    /*reuse_a=*/false, /*reuse_b=*/false);
    }

    // D layout: VGPR r -> M = r + 8*half ; lane -> N = lm. Add bias, store f32.
    #pragma unroll
    for (int mi = 0; mi < 4; ++mi) {
        #pragma unroll
        for (int r = 0; r < 8; ++r) {
            const int m = mbase + mi * 16 + r + half * 8;
            const float bias = pwb[m];
            float* orow = out + ((size_t)b * C_OUT + m) * T_OUT;
            #pragma unroll
            for (int ni = 0; ni < 2; ++ni) {
                const int t = nbase + ni * 16 + lm;
                if (t < T_OUT) orow[t] = acc[mi][ni][r] + bias;
            }
        }
    }
}

extern "C" void kernel_launch(void* const* d_in, const int* in_sizes, int n_in,
                              void* d_out, int out_size, void* d_ws, size_t ws_size,
                              hipStream_t stream) {
    const float* x    = (const float*)d_in[0];  // [B][C_IN][T_IN]
    const float* dw_w = (const float*)d_in[1];  // [C_IN][1][K]
    const float* dw_b = (const float*)d_in[2];  // [C_IN]
    const float* pw_w = (const float*)d_in[3];  // [C_OUT][C_IN]
    const float* pw_b = (const float*)d_in[4];  // [C_OUT]
    float* out = (float*)d_out;                 // [B][C_OUT][T_OUT]

    // workspace layout (all 16B aligned):
    //   [0, 512KB)            : pw_w in bf16
    //   [512KB, 512KB+33.5MB) : Yt[b][t_pad][c] in bf16
    unsigned short* wb = (unsigned short*)d_ws;
    unsigned short* yt = (unsigned short*)((char*)d_ws + (size_t)C_OUT * C_IN * 2);

    // 1) weight conversion
    {
        int n = C_OUT * C_IN;
        cvt_w_kernel<<<(n + 255) / 256, 256, 0, stream>>>(pw_w, wb, n);
    }
    // 2) depthwise conv -> transposed bf16 Y
    {
        dim3 grid(T_PAD / 128, C_IN / 32, BATCH);
        dw_kernel<<<grid, 256, 0, stream>>>(x, dw_w, dw_b, yt);
    }
    // 3) pointwise GEMM with bf16 WMMA
    {
        dim3 grid(T_PAD / 128, C_OUT / 128, BATCH);
        gemm_kernel<<<grid, 256, 0, stream>>>(wb, yt, pw_b, out);
    }
}
